// LSTM_45681272160513
// MI455X (gfx1250) — matrix operand order repaired and checked by reading, not compiled, not run on hardware
//
#include <hip/hip_runtime.h>

// ---------------------------------------------------------------------------
// LSTM forward on MI455X (gfx1250), wave32 + WMMA f16 (fp32 accumulate).
// Phase 1: Xbuf[t,b, 4H] = inputs @ [Wxi|Wxf|Wxo|Wxc] + bias   (big WMMA GEMM)
// Phase 2: persistent kernel, 512 sequential steps with grid barrier:
//          recurrent weights staged in LDS once; G = H_{t-1} @ Wh via WMMA,
//          gates fused in registers; x/C prefetched under the WMMA chain.
// ---------------------------------------------------------------------------

typedef __attribute__((ext_vector_type(16))) _Float16 v16h;
typedef __attribute__((ext_vector_type(8)))  _Float16 v8h;
typedef __attribute__((ext_vector_type(8)))  float    v8f;

#define T_STEPS 512
#define BATCH   64
#define DIM     512
#define HID     512
#define NG      2048              // 4 * HID (gates i,f,o,c concatenated)
#define MTOT    (T_STEPS * BATCH) // 32768
#define NBLK_SEQ 8                // persistent blocks for the recurrence
#define BPITCH  520               // LDS row pitch in halves (bank-conflict pad)
#define LDS_BYTES (256 * BPITCH * 2)   // 266,240 B <= 320 KB WGP LDS

// ---------------- WMMA helpers --------------------------------------------

// A fragment: 16x32 (MxK) f16, row-major source, leading dim ld (elements).
// Lane m=L&15; lanes<16 hold K=0..15, lanes>=16 hold K=16..31
// of the slice. Two contiguous 16B chunks per lane.
__device__ __forceinline__ v16h ldA(const _Float16* __restrict__ base,
                                    int row0, int ld, int k0, int lane) {
  const _Float16* p = base + (size_t)(row0 + (lane & 15)) * ld
                           + (k0 + ((lane >> 4) << 4));
  v8h lo = *(const v8h*)(p);        // K = kh .. kh+7
  v8h hi = *(const v8h*)(p + 8);    // K = kh+8 .. kh+15
  return __builtin_shufflevector(lo, hi,
      0,1,2,3,4,5,6,7,8,9,10,11,12,13,14,15);
}

// B fragment: 32x16 (KxN) f16 from pre-transposed weights Wp[n][k].
// Lane n=L&15; lanes<16 hold K=0..15, lanes>=16 hold K=16..31 of the slice
// -> one contiguous 32B chunk per lane (global path, 32B aligned).
__device__ __forceinline__ v16h ldB(const _Float16* __restrict__ base,
                                    int nrow0, int ld, int k0, int lane) {
  const _Float16* p = base + (size_t)(nrow0 + (lane & 15)) * ld
                           + (k0 + ((lane >> 4) << 4));
  return *(const v16h*)p;
}

// Same B fragment pattern but from LDS with 16B-aligned row pitch (BPITCH):
// two 16B ds loads (rows are only 16B aligned due to the pad).
__device__ __forceinline__ v16h ldB_lds(const _Float16* p) {
  v8h lo = *(const v8h*)(p);
  v8h hi = *(const v8h*)(p + 8);
  return __builtin_shufflevector(lo, hi,
      0,1,2,3,4,5,6,7,8,9,10,11,12,13,14,15);
}

__device__ __forceinline__ v8f wmma16(v16h a, v16h b, v8f c) {
  return __builtin_amdgcn_wmma_f32_16x16x32_f16(
      /*neg_a=*/false, a, /*neg_b=*/false, b,
      /*c_mod=*/(short)0, c, /*reuse_a=*/false, /*reuse_b=*/false);
}

__device__ __forceinline__ float sigmoid_f(float x) {
  return 1.0f / (1.0f + __expf(-x));
}
__device__ __forceinline__ float tanh_f(float x) {
  float e = __expf(2.0f * x);
  return 1.0f - 2.0f / (e + 1.0f);   // stable at +/-inf
}

// ---------------- prep kernels --------------------------------------------

__global__ void f32_to_f16_k(const float* __restrict__ s,
                             _Float16* __restrict__ d, int n) {
  int i = blockIdx.x * blockDim.x + threadIdx.x;
  if (i < n) d[i] = (_Float16)s[i];
}

// dst[n][k] = W_g[k][h] fp16, n = g*512 + h  (weights transposed for B frags)
__global__ void pack_w_k(const float* __restrict__ W0, const float* __restrict__ W1,
                         const float* __restrict__ W2, const float* __restrict__ W3,
                         _Float16* __restrict__ dst) {
  int e = blockIdx.x * blockDim.x + threadIdx.x;
  if (e >= NG * DIM) return;
  int nrow = e >> 9;          // /512
  int k    = e & 511;
  int g    = nrow >> 9;
  int h    = nrow & 511;
  const float* W = (g == 0) ? W0 : (g == 1) ? W1 : (g == 2) ? W2 : W3;
  dst[e] = (_Float16)W[(size_t)k * HID + h];
}

__global__ void pack_b_k(const float* __restrict__ b0, const float* __restrict__ b1,
                         const float* __restrict__ b2, const float* __restrict__ b3,
                         float* __restrict__ dst) {
  int n = blockIdx.x * blockDim.x + threadIdx.x;
  if (n >= NG) return;
  int g = n >> 9, h = n & 511;
  const float* b = (g == 0) ? b0 : (g == 1) ? b1 : (g == 2) ? b2 : b3;
  dst[n] = b[h];
}

__global__ void zero_k(unsigned* __restrict__ p, int nwords) {
  int i = blockIdx.x * blockDim.x + threadIdx.x;
  if (i < nwords) p[i] = 0u;
}

// ---------------- Phase 1: input-projection GEMM --------------------------
// C[32768][2048] = A[32768][512] * Bp^T + bias ; one wave = 64x64 tile block.

__global__ void __launch_bounds__(256) xproj_gemm(
    const _Float16* __restrict__ A,    // [MTOT][DIM] fp16
    const _Float16* __restrict__ Bp,   // [NG][DIM]  fp16 (W^T packed)
    const float*    __restrict__ bias, // [NG]
    float*          __restrict__ C)    // [MTOT][NG]
{
  const int lane = threadIdx.x & 31;
  const int wave = threadIdx.x >> 5;
  const int job  = blockIdx.x * 8 + wave;   // 16384 jobs total
  const int m0 = (job >> 5) * 64;           // 512 row blocks
  const int n0 = (job & 31) * 64;           // 32 col blocks

  v8f acc[4][4] = {};
  for (int k0 = 0; k0 < DIM; k0 += 32) {
    v16h a[4], b[4];
#pragma unroll
    for (int i = 0; i < 4; ++i) a[i] = ldA(A,  m0 + 16 * i, DIM, k0, lane);
#pragma unroll
    for (int j = 0; j < 4; ++j) b[j] = ldB(Bp, n0 + 16 * j, DIM, k0, lane);
#pragma unroll
    for (int i = 0; i < 4; ++i)
#pragma unroll
      for (int j = 0; j < 4; ++j)
        acc[i][j] = wmma16(a[i], b[j], acc[i][j]);
  }

  // C/D layout: lane = (m_half, n) ; VGPR v -> M = 8*(lane>>4) + v
  const int nn = lane & 15;
  const int mb = (lane >> 4) << 3;
#pragma unroll
  for (int j = 0; j < 4; ++j) {
    const float bv = bias[n0 + 16 * j + nn];
#pragma unroll
    for (int i = 0; i < 4; ++i) {
#pragma unroll
      for (int v = 0; v < 8; ++v)
        C[(size_t)(m0 + 16 * i + mb + v) * NG + (n0 + 16 * j + nn)]
            = acc[i][j][v] + bv;
    }
  }
}

// ---------------- Phase 2: persistent recurrent kernel --------------------
// 8 blocks x 256 threads = 64 waves. wave w: h-tile = w>>1 (16 hidden cols),
// m-half = w&1 (batch rows 0..31 / 32..63). Each wave computes its h-tile
// for ALL FOUR gates -> gate math fully fused in registers. The block's
// slice of Wh^T (4 gates x 64 h-cols x 512 k fp16 = 256 KB) is staged in
// LDS once; per-step B fragments come from ds loads off the critical path.

__global__ void __launch_bounds__(256) lstm_seq(
    const _Float16* __restrict__ Whp,  // [NG][HID] fp16 (Wh^T packed)
    const float*    __restrict__ Xbuf, // [MTOT][NG] fp32 (x-projections + bias)
    _Float16*       __restrict__ Hbuf, // [2][BATCH][HID] fp16 (double buffer)
    float*          __restrict__ Cbuf, // [BATCH][HID] fp32
    float*          __restrict__ outs, // [T][BATCH][HID]
    float*          __restrict__ Hf,   // [BATCH][HID]
    float*          __restrict__ Cf,   // [BATCH][HID]
    unsigned*       __restrict__ cnt)  // grid-barrier counter (monotonic)
{
  extern __shared__ _Float16 Bs[];     // [256][BPITCH]

  const int tid  = threadIdx.x;
  const int lane = tid & 31;
  const int wave = tid >> 5;
  const int w    = blockIdx.x * 8 + wave;  // 0..63
  const int htl  = wave >> 1;              // 0..3  local hidden tile
  const int mh   = w & 1;                  // batch half
  const int nn   = lane & 15;
  const int mb   = (lane >> 4) << 3;
  const int hcol = blockIdx.x * 64 + htl * 16 + nn;   // global hidden column

  // ---- stage this block's Wh^T slice into LDS (once) ----
  // rows: g*64 + hr  <-  Whp[g*HID + blockIdx.x*64 + hr][k], 16B chunks,
  // coalesced across lanes within a row.
  for (int idx = tid; idx < 256 * (HID / 8); idx += 256) {
    const int row   = idx >> 6;        // 64 chunks per row
    const int chunk = idx & 63;
    const int g  = row >> 6;
    const int hr = row & 63;
    const _Float16* src =
        Whp + (size_t)(g * HID + blockIdx.x * 64 + hr) * HID + chunk * 8;
    *(v8h*)&Bs[row * BPITCH + chunk * 8] = *(const v8h*)src;
  }
  __syncthreads();

  // per-lane LDS base for B fragments of each gate
  const _Float16* bls[4];
#pragma unroll
  for (int g = 0; g < 4; ++g)
    bls[g] = &Bs[(g * 64 + htl * 16 + (lane & 15)) * BPITCH + ((lane >> 4) << 4)];

  for (int t = 0; t < T_STEPS; ++t) {
    const _Float16* Hin  = Hbuf + (size_t)( t      & 1) * (BATCH * HID);
    _Float16*       Hout = Hbuf + (size_t)((t + 1) & 1) * (BATCH * HID);

    // ---- prefetch x-projections and C state for this step (HBM/L2 latency
    //      hidden under the WMMA chain below) ----
    const float* xb = Xbuf + (size_t)t * BATCH * NG;
    float xv[2][4][8];
    float cv[2][8];
#pragma unroll
    for (int i = 0; i < 2; ++i) {
#pragma unroll
      for (int v = 0; v < 8; ++v) {
        const int brow = mh * 32 + i * 16 + mb + v;
        const float* xr = xb + (size_t)brow * NG;
#pragma unroll
        for (int g = 0; g < 4; ++g) xv[i][g][v] = xr[g * HID + hcol];
        cv[i][v] = Cbuf[brow * HID + hcol];
      }
    }

    // ---- recurrent GEMM: acc[m-tile][gate] += H_{t-1} x Wh ----
    v8f acc[2][4] = {};
    for (int k0 = 0; k0 < HID; k0 += 32) {
      v16h a0 = ldA(Hin, mh * 32,      HID, k0, lane);
      v16h a1 = ldA(Hin, mh * 32 + 16, HID, k0, lane);
#pragma unroll
      for (int g = 0; g < 4; ++g) {
        v16h b = ldB_lds(bls[g] + k0);
        acc[0][g] = wmma16(a0, b, acc[0][g]);
        acc[1][g] = wmma16(a1, b, acc[1][g]);
      }
    }

    // ---- fused gate math + state update ----
#pragma unroll
    for (int i = 0; i < 2; ++i) {
#pragma unroll
      for (int v = 0; v < 8; ++v) {
        const int brow = mh * 32 + i * 16 + mb + v;
        const float gi = sigmoid_f(acc[i][0][v] + xv[i][0][v]);
        const float gf = sigmoid_f(acc[i][1][v] + xv[i][1][v]);
        const float go = sigmoid_f(acc[i][2][v] + xv[i][2][v]);
        const float gc = tanh_f   (acc[i][3][v] + xv[i][3][v]);
        const int ch = brow * HID + hcol;
        const float c = gf * cv[i][v] + gi * gc;
        const float h = go * tanh_f(c);
        Cbuf[ch] = c;
        outs[(size_t)t * BATCH * HID + ch] = h;
        Hout[ch] = (_Float16)h;
        if (t == T_STEPS - 1) { Hf[ch] = h; Cf[ch] = c; }
      }
    }

    // ---- grid barrier (monotonic counter, agent scope) ----
    __syncthreads();
    if (tid == 0) {
      __hip_atomic_fetch_add(cnt, 1u, __ATOMIC_RELEASE, __HIP_MEMORY_SCOPE_AGENT);
      const unsigned target = (unsigned)(t + 1) * NBLK_SEQ;
      while (__hip_atomic_load(cnt, __ATOMIC_ACQUIRE, __HIP_MEMORY_SCOPE_AGENT)
             < target) {
        __builtin_amdgcn_s_sleep(2);
      }
    }
    __syncthreads();
  }
}

// ---------------- host entry ----------------------------------------------

extern "C" void kernel_launch(void* const* d_in, const int* in_sizes, int n_in,
                              void* d_out, int out_size, void* d_ws, size_t ws_size,
                              hipStream_t stream) {
  (void)in_sizes; (void)n_in; (void)out_size; (void)ws_size;

  const float* inputs = (const float*)d_in[0];
  const float* W_xi = (const float*)d_in[1];
  const float* W_hi = (const float*)d_in[2];
  const float* b_i  = (const float*)d_in[3];
  const float* W_xf = (const float*)d_in[4];
  const float* W_hf = (const float*)d_in[5];
  const float* b_f  = (const float*)d_in[6];
  const float* W_xo = (const float*)d_in[7];
  const float* W_ho = (const float*)d_in[8];
  const float* b_o  = (const float*)d_in[9];
  const float* W_xc = (const float*)d_in[10];
  const float* W_hc = (const float*)d_in[11];
  const float* b_c  = (const float*)d_in[12];

  char* ws = (char*)d_ws;
  size_t off = 0;
  auto alloc = [&](size_t bytes) -> char* {
    char* p = ws + off;
    off = (off + bytes + 255) & ~(size_t)255;
    return p;
  };
  _Float16* Ain  = (_Float16*)alloc((size_t)MTOT * DIM * 2);   // 32 MB
  _Float16* Wxp  = (_Float16*)alloc((size_t)NG * DIM * 2);     // 2 MB
  _Float16* Whp  = (_Float16*)alloc((size_t)NG * HID * 2);     // 2 MB
  float*    bp   = (float*)   alloc((size_t)NG * 4);
  _Float16* Hbuf = (_Float16*)alloc((size_t)2 * BATCH * HID * 2);
  float*    Cbuf = (float*)   alloc((size_t)BATCH * HID * 4);
  unsigned* cnt  = (unsigned*)alloc(256);
  float*    Xbuf = (float*)   alloc((size_t)MTOT * NG * 4);    // 256 MB

  float* outs = (float*)d_out;
  float* Hf   = outs + (size_t)T_STEPS * BATCH * HID;
  float* Cf   = Hf + (size_t)BATCH * HID;

  // prep: convert inputs to fp16, pack/transpose weights, pack bias, zero state
  {
    const int n = MTOT * DIM;
    f32_to_f16_k<<<(n + 255) / 256, 256, 0, stream>>>(inputs, Ain, n);
  }
  pack_w_k<<<(NG * DIM + 255) / 256, 256, 0, stream>>>(W_xi, W_xf, W_xo, W_xc, Wxp);
  pack_w_k<<<(NG * HID + 255) / 256, 256, 0, stream>>>(W_hi, W_hf, W_ho, W_hc, Whp);
  pack_b_k<<<(NG + 255) / 256, 256, 0, stream>>>(b_i, b_f, b_o, b_c, bp);
  {
    const int hw = 2 * BATCH * HID * 2 / 4;   // Hbuf words
    zero_k<<<(hw + 255) / 256, 256, 0, stream>>>((unsigned*)Hbuf, hw);
    const int cw = BATCH * HID;               // Cbuf words
    zero_k<<<(cw + 255) / 256, 256, 0, stream>>>((unsigned*)Cbuf, cw);
    zero_k<<<1, 64, 0, stream>>>(cnt, 64);
  }

  // Phase 1: input projections (one big WMMA GEMM)
  xproj_gemm<<<2048, 256, 0, stream>>>(Ain, Wxp, bp, Xbuf);

  // Phase 2: persistent recurrence with per-step grid barrier, Wh in LDS
  lstm_seq<<<NBLK_SEQ, 256, LDS_BYTES, stream>>>(Whp, Xbuf, Hbuf, Cbuf,
                                                 outs, Hf, Cf, cnt);
}
